// DeepGraphSAGE_70497593197183
// MI455X (gfx1250) — matrix operand order repaired and checked by reading, not compiled, not run on hardware
//
#include <hip/hip_runtime.h>

// DeepGraphSAGE for MI455X (gfx1250, wave32).
// fp32 aggregation/atomics + bf16 WMMA GEMMs (v_wmma_f32_16x16x32_bf16).

typedef __bf16 bf16_t;
typedef __attribute__((ext_vector_type(8)))  __bf16 v8bf;
typedef __attribute__((ext_vector_type(16))) __bf16 v16bf;
typedef __attribute__((ext_vector_type(8)))  float  v8f;

#define TPB 256
#define LDSP 136   // LDS B-tile row pitch in bf16 elems: 272B (16B aligned, 4-bank stagger/row)

static inline unsigned cdiv_u(long long a, long long b) { return (unsigned)((a + b - 1) / b); }

// ---------------- small utility kernels ----------------

__global__ void dgs_count_deg(const int* __restrict__ dst, float* __restrict__ deg, int E) {
    int e = blockIdx.x * blockDim.x + threadIdx.x;
    if (e < E) atomicAdd(&deg[dst[e]], 1.0f);
}

__global__ void dgs_count_sz(const int* __restrict__ batch, float* __restrict__ sz, int N) {
    int n = blockIdx.x * blockDim.x + threadIdx.x;
    if (n < N) atomicAdd(&sz[batch[n]], 1.0f);
}

__global__ void dgs_recip_max1(float* __restrict__ p, int n) {
    int i = blockIdx.x * blockDim.x + threadIdx.x;
    if (i < n) p[i] = 1.0f / fmaxf(p[i], 1.0f);
}

// f32 [rowsV x colsV] -> bf16 [rowsP x colsP], zero padded
__global__ void dgs_conv_pad(const float* __restrict__ src, bf16_t* __restrict__ dst,
                             int rowsV, int colsV, long long totalP, int colsP) {
    long long idx = (long long)blockIdx.x * blockDim.x + threadIdx.x;
    if (idx >= totalP) return;
    int r = (int)(idx / colsP);
    int c = (int)(idx - (long long)r * colsP);
    float v = (r < rowsV && c < colsV) ? src[(long long)r * colsV + c] : 0.0f;
    dst[idx] = (bf16_t)v;
}

// Agg[dst] += X[src], edge-parallel, float2 granularity (Fin always even: 50/512)
__global__ void dgs_scatter(const float* __restrict__ X, const int* __restrict__ src,
                            const int* __restrict__ dst, float* __restrict__ Agg,
                            long long total, int half, int Fin) {
    long long tid = (long long)blockIdx.x * blockDim.x + threadIdx.x;
    if (tid >= total) return;
    int e = (int)(tid / half);
    int c = (int)(tid - (long long)e * half) * 2;
    int s = src[e], d = dst[e];
    const float2 v = *(const float2*)(X + (size_t)s * Fin + c);
    atomicAdd(&Agg[(size_t)d * Fin + c + 0], v.x);
    atomicAdd(&Agg[(size_t)d * Fin + c + 1], v.y);
}

// out[n][k] = bf16(Agg[n][k] * invdeg[n]) for k < Fin, else 0   (out stride Kpad)
__global__ void dgs_scale_conv(const float* __restrict__ Agg, const float* __restrict__ invdeg,
                               bf16_t* __restrict__ out, long long total, int Fin, int Kpad) {
    long long idx = (long long)blockIdx.x * blockDim.x + threadIdx.x;
    if (idx >= total) return;
    int n = (int)(idx / Kpad);
    int k = (int)(idx - (long long)n * Kpad);
    float v = (k < Fin) ? Agg[(size_t)n * Fin + k] * invdeg[n] : 0.0f;
    out[idx] = (bf16_t)v;
}

// ---------------- GraphNorm ----------------

__global__ void dgs_gn_stats(const float* __restrict__ Hm, const int* __restrict__ batch,
                             float* __restrict__ S1, float* __restrict__ S2,
                             long long total, int C) {
    long long idx = (long long)blockIdx.x * blockDim.x + threadIdx.x;
    if (idx >= total) return;
    int n = (int)(idx / C);
    int c = (int)(idx - (long long)n * C);
    float h = Hm[idx];
    int g = batch[n];
    atomicAdd(&S1[(size_t)g * C + c], h);
    atomicAdd(&S2[(size_t)g * C + c], h * h);
}

// shift = a*mean ; invstd = rsqrt(E[x^2] - (2a - a^2)*mean^2 + eps)
__global__ void dgs_gn_finalize(const float* __restrict__ S1, const float* __restrict__ S2,
                                const float* __restrict__ invsz, const float* __restrict__ alpha,
                                float* __restrict__ shift, float* __restrict__ istd,
                                int total, int C) {
    int idx = blockIdx.x * blockDim.x + threadIdx.x;
    if (idx >= total) return;
    int g = idx / C;
    int c = idx - g * C;
    float is   = invsz[g];
    float mean = S1[idx] * is;
    float ex2  = S2[idx] * is;
    float a    = alpha[c];
    float var  = ex2 - (2.0f * a - a * a) * mean * mean;
    shift[idx] = a * mean;
    istd[idx]  = rsqrtf(var + 1e-5f);
}

// Hm = relu((Hm - shift)*istd*w + b), also writes bf16 copy for next GEMM
__global__ void dgs_gn_apply(float* __restrict__ Hm, const int* __restrict__ batch,
                             const float* __restrict__ shift, const float* __restrict__ istd,
                             const float* __restrict__ w, const float* __restrict__ b,
                             bf16_t* __restrict__ XB, long long total, int C) {
    long long idx = (long long)blockIdx.x * blockDim.x + threadIdx.x;
    if (idx >= total) return;
    int n = (int)(idx / C);
    int c = (int)(idx - (long long)n * C);
    int g = batch[n];
    size_t gi = (size_t)g * C + c;
    float v = (Hm[idx] - shift[gi]) * istd[gi] * w[c] + b[c];
    v = fmaxf(v, 0.0f);
    Hm[idx] = v;
    XB[idx] = (bf16_t)v;
}

// ---------------- dual-source WMMA GEMM ----------------
// Out[m][o] = sum_k A1[m][k]*B1[o][k] + sum_k A2[m][k]*B2[o][k] + bias[o]
// A: [M x K] bf16 row-major, B: [FoutP x K] bf16 row-major (weight layout, no transpose needed)
// block = 256 thr (8 waves); wave tile 16x64 (4 wmma accumulators); block tile 128x64.
__global__ __launch_bounds__(TPB)
void dgs_gemm_dual(const bf16_t* __restrict__ A1, const bf16_t* __restrict__ B1,
                   const bf16_t* __restrict__ A2, const bf16_t* __restrict__ B2,
                   const float* __restrict__ bias, float* __restrict__ Out,
                   int M, int K, int Fout) {
    __shared__ bf16_t lb1[64 * LDSP];
    __shared__ bf16_t lb2[64 * LDSP];

    const int wave = threadIdx.x >> 5;
    const int lane = threadIdx.x & 31;
    const int half = lane >> 4;   // 0 or 1
    const int lrow = lane & 15;
    const int m0 = blockIdx.x * 128 + wave * 16;
    const int n0 = blockIdx.y * 64;

    v8f acc[4] = {};

    int arow = m0 + lrow;
    if (arow >= M) arow = M - 1;                    // clamp; stores are guarded
    const bf16_t* a1p = A1 + (size_t)arow * K;
    const bf16_t* a2p = A2 + (size_t)arow * K;

    for (int kc = 0; kc < K; kc += 128) {
        const int chunk = (K - kc < 128) ? (K - kc) : 128;   // K is a multiple of 32 (64 or 512)
        const int cpv = chunk >> 3;                          // 8-elem vectors per B row
        __syncthreads();
        // stage B1/B2 chunk [64 rows x chunk] into LDS (all 256 threads)
        for (int idx = threadIdx.x; idx < 64 * cpv; idx += TPB) {
            const int r  = idx / cpv;
            const int c8 = (idx - r * cpv) << 3;
            const size_t go = (size_t)(n0 + r) * K + kc + c8;
            *(v8bf*)(&lb1[r * LDSP + c8]) = *(const v8bf*)(B1 + go);
            *(v8bf*)(&lb2[r * LDSP + c8]) = *(const v8bf*)(B2 + go);
        }
        __syncthreads();

        for (int kin = 0; kin < chunk; kin += 32) {
            // A fragment per ISA layout: lane<16 holds K{0..7,16..23}, lane>=16 K{8..15,24..31}
            const int kA = kc + kin + half * 8;
            union { v16bf v; v8bf h[2]; } a;
            a.h[0] = *(const v8bf*)(a1p + kA);
            a.h[1] = *(const v8bf*)(a1p + kA + 16);
#pragma unroll
            for (int t = 0; t < 4; ++t) {
                // B fragment: same K order as A: lane<16 K{0..7,16..23}, lane>=16 K{8..15,24..31}
                const bf16_t* bp = &lb1[(t * 16 + lrow) * LDSP + kin + half * 8];
                union { v16bf v; v8bf h[2]; } b;
                b.h[0] = *(const v8bf*)(bp);
                b.h[1] = *(const v8bf*)(bp + 16);
                acc[t] = __builtin_amdgcn_wmma_f32_16x16x32_bf16(
                    false, a.v, false, b.v, (short)0, acc[t], false, false);
            }
            a.h[0] = *(const v8bf*)(a2p + kA);
            a.h[1] = *(const v8bf*)(a2p + kA + 16);
#pragma unroll
            for (int t = 0; t < 4; ++t) {
                const bf16_t* bp = &lb2[(t * 16 + lrow) * LDSP + kin + half * 8];
                union { v16bf v; v8bf h[2]; } b;
                b.h[0] = *(const v8bf*)(bp);
                b.h[1] = *(const v8bf*)(bp + 16);
                acc[t] = __builtin_amdgcn_wmma_f32_16x16x32_bf16(
                    false, a.v, false, b.v, (short)0, acc[t], false, false);
            }
        }
    }

    // C/D layout: element(m_l, n_l) -> vgpr m_l%8, lane (m_l/8)*16 + n_l
#pragma unroll
    for (int t = 0; t < 4; ++t) {
        const int o = n0 + t * 16 + lrow;
        if (o >= Fout) continue;
        const float bv = bias[o];
#pragma unroll
        for (int v = 0; v < 8; ++v) {
            const int m = m0 + half * 8 + v;
            if (m < M) Out[(size_t)m * Fout + o] = acc[t][v] + bv;
        }
    }
}

// ---------------- host orchestration ----------------

extern "C" void kernel_launch(void* const* d_in, const int* in_sizes, int n_in,
                              void* d_out, int out_size, void* d_ws, size_t ws_size,
                              hipStream_t stream) {
    const float* x     = (const float*)d_in[0];
    const int*   ei    = (const int*)d_in[1];
    const int*   batch = (const int*)d_in[2];

    const int N = in_sizes[2];
    const int E = in_sizes[1] / 2;
    const int F = in_sizes[0] / N;       // 50
    const int H = in_sizes[5];           // 512 (b1)
    const int O = in_sizes[17];          // 121 (b5)
    const int G = 20;                    // graphs (matches reference)
    const int KP1 = ((F + 63) / 64) * 64;   // 64
    const int OP  = ((O + 63) / 64) * 64;   // 128
    const int* srcp = ei;
    const int* dstp = ei + E;

    // ---- carve workspace (~413 MB total) ----
    char* base = (char*)d_ws;
    size_t off = 0;
    auto alloc = [&](size_t bytes) -> char* {
        char* p = base + off;
        off = (off + bytes + 255) & ~((size_t)255);
        return p;
    };
    float*  P0   = (float*)alloc((size_t)N * H * sizeof(float));
    float*  P1   = (float*)alloc((size_t)N * H * sizeof(float));
    float*  Agg  = (float*)alloc((size_t)N * H * sizeof(float));
    bf16_t* XB   = (bf16_t*)alloc((size_t)N * H * sizeof(bf16_t));
    bf16_t* AggB = (bf16_t*)alloc((size_t)N * H * sizeof(bf16_t));
    bf16_t* WlB[5];
    bf16_t* WrB[5];
    WlB[0] = (bf16_t*)alloc((size_t)H * KP1 * sizeof(bf16_t));
    WrB[0] = (bf16_t*)alloc((size_t)H * KP1 * sizeof(bf16_t));
    for (int i = 1; i < 4; ++i) {
        WlB[i] = (bf16_t*)alloc((size_t)H * H * sizeof(bf16_t));
        WrB[i] = (bf16_t*)alloc((size_t)H * H * sizeof(bf16_t));
    }
    WlB[4] = (bf16_t*)alloc((size_t)OP * H * sizeof(bf16_t));
    WrB[4] = (bf16_t*)alloc((size_t)OP * H * sizeof(bf16_t));
    float* deg   = (float*)alloc((size_t)N * sizeof(float));   // becomes inv_deg
    float* invsz = (float*)alloc((size_t)G * sizeof(float));   // becomes inv_sz
    float* S1    = (float*)alloc((size_t)G * H * sizeof(float));
    float* S2    = (float*)alloc((size_t)G * H * sizeof(float));
    float* Shift = (float*)alloc((size_t)G * H * sizeof(float));
    float* Istd  = (float*)alloc((size_t)G * H * sizeof(float));
    (void)ws_size; (void)n_in; (void)out_size;

    // ---- degree / graph sizes ----
    hipMemsetAsync(deg, 0, (size_t)N * sizeof(float), stream);
    hipMemsetAsync(invsz, 0, (size_t)G * sizeof(float), stream);
    dgs_count_deg<<<cdiv_u(E, TPB), TPB, 0, stream>>>(dstp, deg, E);
    dgs_count_sz<<<cdiv_u(N, TPB), TPB, 0, stream>>>(batch, invsz, N);
    dgs_recip_max1<<<cdiv_u(N, TPB), TPB, 0, stream>>>(deg, N);
    dgs_recip_max1<<<1, TPB, 0, stream>>>(invsz, G);

    // ---- convert weights to padded bf16 ----
    for (int i = 0; i < 5; ++i) {
        const float* wl = (const float*)d_in[3 + 3 * i];
        const float* wr = (const float*)d_in[4 + 3 * i];
        const int rowsV = (i == 4) ? O : H;
        const int colsV = (i == 0) ? F : H;
        const int rowsP = (i == 4) ? OP : H;
        const int colsP = (i == 0) ? KP1 : H;
        const long long tot = (long long)rowsP * colsP;
        dgs_conv_pad<<<cdiv_u(tot, TPB), TPB, 0, stream>>>(wl, WlB[i], rowsV, colsV, tot, colsP);
        dgs_conv_pad<<<cdiv_u(tot, TPB), TPB, 0, stream>>>(wr, WrB[i], rowsV, colsV, tot, colsP);
    }
    // x -> bf16 padded [N x 64] for layer-1 self path
    {
        const long long tot = (long long)N * KP1;
        dgs_conv_pad<<<cdiv_u(tot, TPB), TPB, 0, stream>>>(x, XB, N, F, tot, KP1);
    }

    // ---- 5 SAGE layers ----
    const float* Xcur = x;          // fp32 features, stride Fin
    float* Pbuf[2] = {P0, P1};
    for (int layer = 0; layer < 5; ++layer) {
        const int Fin = (layer == 0) ? F : H;
        const int Kp  = (layer == 0) ? KP1 : H;
        const int Fo  = (layer == 4) ? O : H;
        const int FoP = (layer == 4) ? OP : H;

        // mean-aggregate neighbors
        hipMemsetAsync(Agg, 0, (size_t)N * Fin * sizeof(float), stream);
        const long long st = (long long)E * (Fin / 2);
        dgs_scatter<<<cdiv_u(st, TPB), TPB, 0, stream>>>(Xcur, srcp, dstp, Agg, st, Fin / 2, Fin);
        const long long sc = (long long)N * Kp;
        dgs_scale_conv<<<cdiv_u(sc, TPB), TPB, 0, stream>>>(Agg, deg, AggB, sc, Fin, Kp);

        // dual GEMM: Agg@Wl^T + X@Wr^T + b
        float* Out = (layer == 4) ? (float*)d_out : Pbuf[layer & 1];
        dim3 gg(cdiv_u(N, 128), FoP / 64);
        dgs_gemm_dual<<<gg, TPB, 0, stream>>>(AggB, WlB[layer], XB, WrB[layer],
                                              (const float*)d_in[5 + 3 * layer],
                                              Out, N, Kp, Fo);

        if (layer < 4) {
            // GraphNorm + ReLU (fused bf16 export for next GEMM)
            hipMemsetAsync(S1, 0, (size_t)G * H * sizeof(float), stream);
            hipMemsetAsync(S2, 0, (size_t)G * H * sizeof(float), stream);
            const long long tot = (long long)N * H;
            dgs_gn_stats<<<cdiv_u(tot, TPB), TPB, 0, stream>>>(Out, batch, S1, S2, tot, H);
            dgs_gn_finalize<<<cdiv_u((long long)G * H, TPB), TPB, 0, stream>>>(
                S1, S2, invsz, (const float*)d_in[20 + 3 * layer], Shift, Istd, G * H, H);
            dgs_gn_apply<<<cdiv_u(tot, TPB), TPB, 0, stream>>>(
                Out, batch, Shift, Istd,
                (const float*)d_in[18 + 3 * layer], (const float*)d_in[19 + 3 * layer],
                XB, tot, H);
            Xcur = Out;
        }
    }
}
